// MoELayer_52956946759945
// MI455X (gfx1250) — matrix operand order repaired and checked by reading, not compiled, not run on hardware
//
#include <hip/hip_runtime.h>

// ---------------------------------------------------------------------------
// MoE dual-expert SwiGLU MLP for MI455X (gfx1250), bf16 WMMA path.
//   out = mask0 * down0(silu(x Wg0) * (x Wu0)) + mask1 * down1(...)
// Shapes: x [8192,1024] f32, Wg/Wu [1024,3584], Wd [3584,1024], out [8192,1024]
// Double-buffered LDS tiles: one barrier per K-step, global loads of tile k+1
// overlap WMMAs on tile k.
// ---------------------------------------------------------------------------

#define D_MODEL 1024
#define D_FF    3584
#define NTOK    8192   // B*S = 2*4096

typedef __attribute__((ext_vector_type(16))) __bf16 bf16x16;
typedef __attribute__((ext_vector_type(8)))  __bf16 bf16x8;
typedef __attribute__((ext_vector_type(8)))  float  f32x8;

union FragAB {
  bf16x16 v;
  bf16x8  h[2];
};

__device__ inline f32x8 wmma_bf16(bf16x16 a, bf16x16 b, f32x8 c) {
  // D = A(16x32 bf16) * B(32x16 bf16) + C(16x16 f32)
  return __builtin_amdgcn_wmma_f32_16x16x32_bf16(
      /*neg_a=*/false, a, /*neg_b=*/false, b,
      /*c_mod=*/(short)0, c, /*reuse_a=*/false, /*reuse_b=*/false);
}

// ---------------------------------------------------------------------------
// Prologue: f32 -> bf16 cast of activations (vectorized x4).
// ---------------------------------------------------------------------------
__global__ __launch_bounds__(256) void cast_x_kernel(const float* __restrict__ x,
                                                     __bf16* __restrict__ xb,
                                                     int n4) {
  int i = blockIdx.x * 256 + threadIdx.x;
  if (i >= n4) return;
  float4 f = ((const float4*)x)[i];
  union { __bf16 b[4]; unsigned long long u; } o;
  o.b[0] = (__bf16)f.x; o.b[1] = (__bf16)f.y;
  o.b[2] = (__bf16)f.z; o.b[3] = (__bf16)f.w;
  ((unsigned long long*)xb)[i] = o.u;
}

// ---------------------------------------------------------------------------
// Prologue: transpose + cast  src[R,C] f32 -> dst[C,R] bf16  (32x32 LDS tile).
// R, C are multiples of 32.
// ---------------------------------------------------------------------------
__global__ __launch_bounds__(256) void transpose_cast_kernel(const float* __restrict__ src,
                                                             __bf16* __restrict__ dst,
                                                             int R, int C) {
  __shared__ float tile[32][33];
  const int tx = threadIdx.x & 31;
  const int ty = threadIdx.x >> 5;           // 0..7
  const int c0 = blockIdx.x * 32;
  const int r0 = blockIdx.y * 32;
#pragma unroll
  for (int i = 0; i < 4; ++i)
    tile[ty + 8 * i][tx] = src[(size_t)(r0 + ty + 8 * i) * C + (c0 + tx)];
  __syncthreads();
#pragma unroll
  for (int i = 0; i < 4; ++i)
    dst[(size_t)(c0 + ty + 8 * i) * R + (r0 + tx)] = (__bf16)tile[tx][ty + 8 * i];
}

// ---------------------------------------------------------------------------
// Fused gate+up GEMM with SiLU epilogue (double-buffered LDS).
//   H[m,f] = silu(X.Gt[f]) * (X.Ut[f]),  X [NTOK,D_MODEL] bf16,
//   Gt/Ut [D_FF, D_MODEL] bf16 (pre-transposed, K-major rows).
// Block: 256 threads = 8 waves as 2(M) x 4(N). BM=64 BN=128 BK=32.
// Each wave owns a 32x32 output patch (2x2 WMMA tiles) for BOTH matrices.
// ---------------------------------------------------------------------------
#define LDSPITCH 40   // 32 bf16 payload + 8 pad; 80B row, keeps 16B alignment

__global__ __launch_bounds__(256) void gemm_gateup_kernel(
    const __bf16* __restrict__ X,
    const __bf16* __restrict__ Gt,
    const __bf16* __restrict__ Ut,
    __bf16* __restrict__ H) {
  __shared__ __align__(16) __bf16 As[2][64 * LDSPITCH];
  __shared__ __align__(16) __bf16 Bg[2][128 * LDSPITCH];
  __shared__ __align__(16) __bf16 Bu[2][128 * LDSPITCH];

  const int t    = threadIdx.x;
  const int lane = t & 31;
  const int w    = t >> 5;
  const int wm   = w >> 2;        // 0..1
  const int wn   = w & 3;         // 0..3
  const int m0   = blockIdx.y * 64;
  const int f0   = blockIdx.x * 128;

  f32x8 accg[2][2], accu[2][2];
#pragma unroll
  for (int a = 0; a < 2; ++a)
#pragma unroll
    for (int b = 0; b < 2; ++b) { accg[a][b] = {}; accu[a][b] = {}; }

  const int lr   = t >> 2;        // 0..63  loader row
  const int lc   = (t & 3) * 8;   // bf16 column chunk (16B)
  const int hsel = lane >> 4;     // half-wave select
  const int rr   = lane & 15;     // row/col within tile
  const int sA   = lr * LDSPITCH + lc;
  const int sB1  = lr * LDSPITCH + lc;
  const int sB2  = (lr + 64) * LDSPITCH + lc;

  const size_t gA  = (size_t)(m0 + lr) * D_MODEL + lc;
  const size_t gB1 = (size_t)(f0 + lr) * D_MODEL + lc;
  const size_t gB2 = (size_t)(f0 + lr + 64) * D_MODEL + lc;

  uint4 ra, rg0, rg1, ru0, ru1;

  // prologue: tile 0 -> LDS buffer 0
  ra  = *(const uint4*)&X [gA];
  rg0 = *(const uint4*)&Gt[gB1];
  rg1 = *(const uint4*)&Gt[gB2];
  ru0 = *(const uint4*)&Ut[gB1];
  ru1 = *(const uint4*)&Ut[gB2];
  *(uint4*)&As[0][sA]  = ra;
  *(uint4*)&Bg[0][sB1] = rg0;
  *(uint4*)&Bg[0][sB2] = rg1;
  *(uint4*)&Bu[0][sB1] = ru0;
  *(uint4*)&Bu[0][sB2] = ru1;
  __syncthreads();

  const int NK = D_MODEL / 32;
  for (int kb = 0; kb < NK; ++kb) {
    const int cur  = kb & 1;
    const int nxt  = cur ^ 1;
    const bool more = (kb + 1) < NK;

    if (more) {  // issue next tile's global loads early (overlap with WMMA)
      const int k1 = (kb + 1) * 32;
      ra  = *(const uint4*)&X [gA  + k1];
      rg0 = *(const uint4*)&Gt[gB1 + k1];
      rg1 = *(const uint4*)&Gt[gB2 + k1];
      ru0 = *(const uint4*)&Ut[gB1 + k1];
      ru1 = *(const uint4*)&Ut[gB2 + k1];
    }

    // A fragments: per lane row rr, 16 consecutive K at hsel*16
    FragAB a[2];
#pragma unroll
    for (int mt = 0; mt < 2; ++mt) {
      const __bf16* p = &As[cur][(wm * 32 + mt * 16 + rr) * LDSPITCH + hsel * 16];
      a[mt].h[0] = *(const bf16x8*)p;
      a[mt].h[1] = *(const bf16x8*)(p + 8);
    }
#pragma unroll
    for (int nt = 0; nt < 2; ++nt) {
      // B fragments: per lane column rr, 16 consecutive K at hsel*16
      FragAB bgf, buf;
      const __bf16* pg = &Bg[cur][(wn * 32 + nt * 16 + rr) * LDSPITCH + hsel * 16];
      bgf.h[0] = *(const bf16x8*)pg;
      bgf.h[1] = *(const bf16x8*)(pg + 8);
      const __bf16* pu = &Bu[cur][(wn * 32 + nt * 16 + rr) * LDSPITCH + hsel * 16];
      buf.h[0] = *(const bf16x8*)pu;
      buf.h[1] = *(const bf16x8*)(pu + 8);
#pragma unroll
      for (int mt = 0; mt < 2; ++mt) {
        accg[mt][nt] = wmma_bf16(a[mt].v, bgf.v, accg[mt][nt]);
        accu[mt][nt] = wmma_bf16(a[mt].v, buf.v, accu[mt][nt]);
      }
    }

    if (more) {  // stage next tile into the other buffer
      *(uint4*)&As[nxt][sA]  = ra;
      *(uint4*)&Bg[nxt][sB1] = rg0;
      *(uint4*)&Bg[nxt][sB2] = rg1;
      *(uint4*)&Bu[nxt][sB1] = ru0;
      *(uint4*)&Bu[nxt][sB2] = ru1;
    }
    __syncthreads();
  }

  // Epilogue: h = silu(g) * u, stored bf16.
  const int rowoff = hsel * 8;
  const int col    = rr;
#pragma unroll
  for (int mt = 0; mt < 2; ++mt)
#pragma unroll
    for (int nt = 0; nt < 2; ++nt) {
      const int gf = f0 + wn * 32 + nt * 16 + col;
#pragma unroll
      for (int j = 0; j < 8; ++j) {
        const int gm = m0 + wm * 32 + mt * 16 + rowoff + j;
        float g = accg[mt][nt][j];
        float u = accu[mt][nt][j];
        float s = g / (1.0f + __expf(-g));
        H[(size_t)gm * D_FF + gf] = (__bf16)(s * u);
      }
    }
}

// ---------------------------------------------------------------------------
// Down-projection GEMM + routing-mask combine into fp32 output (double-buffered).
//   expert 0: Out[m,d]  = (routing[m]==0) ? (H.Dt)[m,d] : 0
//   expert 1: Out[m,d] += (routing[m]==1) ? (H.Dt)[m,d] : 0
// H [NTOK,D_FF] bf16, Dt [D_MODEL,D_FF] bf16 (pre-transposed).
// ---------------------------------------------------------------------------
__global__ __launch_bounds__(256) void gemm_down_kernel(
    const __bf16* __restrict__ H,
    const __bf16* __restrict__ Dt,
    const int* __restrict__ routing,
    float* __restrict__ Out,
    int expert) {
  __shared__ __align__(16) __bf16 As[2][64 * LDSPITCH];
  __shared__ __align__(16) __bf16 Bs[2][128 * LDSPITCH];

  const int t    = threadIdx.x;
  const int lane = t & 31;
  const int w    = t >> 5;
  const int wm   = w >> 2;
  const int wn   = w & 3;
  const int m0   = blockIdx.y * 64;
  const int n0   = blockIdx.x * 128;

  f32x8 acc[2][2];
#pragma unroll
  for (int a = 0; a < 2; ++a)
#pragma unroll
    for (int b = 0; b < 2; ++b) acc[a][b] = {};

  const int lr   = t >> 2;
  const int lc   = (t & 3) * 8;
  const int hsel = lane >> 4;
  const int rr   = lane & 15;
  const int sA   = lr * LDSPITCH + lc;
  const int sB1  = lr * LDSPITCH + lc;
  const int sB2  = (lr + 64) * LDSPITCH + lc;

  const size_t gA  = (size_t)(m0 + lr) * D_FF + lc;
  const size_t gB1 = (size_t)(n0 + lr) * D_FF + lc;
  const size_t gB2 = (size_t)(n0 + lr + 64) * D_FF + lc;

  uint4 ra, rb0, rb1;

  ra  = *(const uint4*)&H [gA];
  rb0 = *(const uint4*)&Dt[gB1];
  rb1 = *(const uint4*)&Dt[gB2];
  *(uint4*)&As[0][sA]  = ra;
  *(uint4*)&Bs[0][sB1] = rb0;
  *(uint4*)&Bs[0][sB2] = rb1;
  __syncthreads();

  const int NK = D_FF / 32;
  for (int kb = 0; kb < NK; ++kb) {
    const int cur  = kb & 1;
    const int nxt  = cur ^ 1;
    const bool more = (kb + 1) < NK;

    if (more) {
      const int k1 = (kb + 1) * 32;
      ra  = *(const uint4*)&H [gA  + k1];
      rb0 = *(const uint4*)&Dt[gB1 + k1];
      rb1 = *(const uint4*)&Dt[gB2 + k1];
    }

    FragAB a[2];
#pragma unroll
    for (int mt = 0; mt < 2; ++mt) {
      const __bf16* p = &As[cur][(wm * 32 + mt * 16 + rr) * LDSPITCH + hsel * 16];
      a[mt].h[0] = *(const bf16x8*)p;
      a[mt].h[1] = *(const bf16x8*)(p + 8);
    }
#pragma unroll
    for (int nt = 0; nt < 2; ++nt) {
      FragAB bb;
      const __bf16* pb = &Bs[cur][(wn * 32 + nt * 16 + rr) * LDSPITCH + hsel * 16];
      bb.h[0] = *(const bf16x8*)pb;
      bb.h[1] = *(const bf16x8*)(pb + 8);
#pragma unroll
      for (int mt = 0; mt < 2; ++mt)
        acc[mt][nt] = wmma_bf16(a[mt].v, bb.v, acc[mt][nt]);
    }

    if (more) {
      *(uint4*)&As[nxt][sA]  = ra;
      *(uint4*)&Bs[nxt][sB1] = rb0;
      *(uint4*)&Bs[nxt][sB2] = rb1;
    }
    __syncthreads();
  }

  const int rowoff = hsel * 8;
  const int col    = rr;
#pragma unroll
  for (int mt = 0; mt < 2; ++mt)
#pragma unroll
    for (int j = 0; j < 8; ++j) {
      const int gm = m0 + wm * 32 + mt * 16 + rowoff + j;
      const float keep = (routing[gm] == expert) ? 1.0f : 0.0f;
#pragma unroll
      for (int nt = 0; nt < 2; ++nt) {
        const int gd = n0 + wn * 32 + nt * 16 + col;
        const size_t oi = (size_t)gm * D_MODEL + gd;
        const float v = keep * acc[mt][nt][j];
        if (expert == 0) Out[oi] = v;
        else             Out[oi] += v;
      }
    }
}

// ---------------------------------------------------------------------------
// Host launcher. Workspace layout (bytes):
//   [0)                xb   : NTOK*D_MODEL bf16            (16 MB)
//   [+16M)             wT[6]: each D_FF*D_MODEL bf16       (6 x 7 MB)
//   [+60M)             h    : NTOK*D_FF bf16               (58.7 MB)
// Total ~119.5 MB.
// ---------------------------------------------------------------------------
extern "C" void kernel_launch(void* const* d_in, const int* in_sizes, int n_in,
                              void* d_out, int out_size, void* d_ws, size_t ws_size,
                              hipStream_t stream) {
  const float* x       = (const float*)d_in[0];
  const int*   routing = (const int*)d_in[1];
  const float* wsrc[6] = {(const float*)d_in[2], (const float*)d_in[3],
                          (const float*)d_in[4], (const float*)d_in[5],
                          (const float*)d_in[6], (const float*)d_in[7]};
  float* out = (float*)d_out;

  char* ws = (char*)d_ws;
  __bf16* xb = (__bf16*)ws;
  size_t off = (size_t)NTOK * D_MODEL * 2;
  __bf16* wT[6];
  for (int i = 0; i < 6; ++i) {
    wT[i] = (__bf16*)(ws + off);
    off += (size_t)D_MODEL * D_FF * 2;
  }
  __bf16* h = (__bf16*)(ws + off);

  // 1) cast activations
  {
    int n4 = NTOK * D_MODEL / 4;
    cast_x_kernel<<<(n4 + 255) / 256, 256, 0, stream>>>(x, xb, n4);
  }
  // 2) transpose + cast weights.
  //    idx 0..5 = wg0, wu0, wd0, wg1, wu1, wd1.
  //    gate/up: src [D_MODEL, D_FF] -> dst [D_FF, D_MODEL]
  //    down   : src [D_FF, D_MODEL] -> dst [D_MODEL, D_FF]
  for (int i = 0; i < 6; ++i) {
    const bool is_down = (i == 2 || i == 5);
    const int R = is_down ? D_FF : D_MODEL;
    const int C = is_down ? D_MODEL : D_FF;
    transpose_cast_kernel<<<dim3(C / 32, R / 32), 256, 0, stream>>>(wsrc[i], wT[i], R, C);
  }
  // 3) experts, h buffer reused
  for (int e = 0; e < 2; ++e) {
    gemm_gateup_kernel<<<dim3(D_FF / 128, NTOK / 64), 256, 0, stream>>>(
        xb, wT[3 * e + 0], wT[3 * e + 1], h);
    gemm_down_kernel<<<dim3(D_MODEL / 128, NTOK / 64), 256, 0, stream>>>(
        h, wT[3 * e + 2], routing, out, e);
  }
}
